// PraxisSMEAR_88287347737174
// MI455X (gfx1250) — matrix operand order repaired and checked by reading, not compiled, not run on hardware
//
#include <hip/hip_runtime.h>

typedef __attribute__((ext_vector_type(16))) __bf16       v16bf;
typedef __attribute__((ext_vector_type(8)))  float        v8f;
typedef __attribute__((ext_vector_type(4)))  unsigned int u32x4;
typedef unsigned short ushort_t;

namespace {
constexpr int kB = 4;
constexpr int kS = 2048;
constexpr int kD = 1024;
constexpr int kE = 8;
constexpr int kT = kB * kS;   // 8192 tokens
constexpr int kH = 8 * kD;    // 8192 rows of W_up
constexpr int kF = 4 * kD;    // 4096 GLU width

constexpr int kKC = 32;                 // K-chunk staged per buffer
constexpr int kRowE = kKC + 8;          // LDS row stride in bf16 elems (80 B, padded)
constexpr int kRowB = kRowE * 2;        // 80 bytes
}

__device__ __forceinline__ ushort_t f2bf(float f) {
  unsigned int u = __float_as_uint(f);
  u += 0x7FFFu + ((u >> 16) & 1u);     // round-to-nearest-even
  return (ushort_t)(u >> 16);
}

union FragBits { u32x4 q[2]; v16bf v; };

// A fragment: 16x32 bf16 tile from row-major [rows][ld] (works for LDS or global ptrs).
// lane<16: row m=lane, K = {0..7} and {16..23}; lane>=16: K = {8..15} and {24..31}
__device__ __forceinline__ v16bf load_a_frag(const ushort_t* base, int ld,
                                             int row0, int lane) {
  const int m = lane & 15, kg = lane >> 4;
  const ushort_t* p = base + (row0 + m) * ld + kg * 8;
  FragBits f;
  f.q[0] = *(const u32x4*)(p);        // 8 bf16 = 16 B
  f.q[1] = *(const u32x4*)(p + 16);   // K + 16 elements
  return f.v;
}

// B fragment: 32x16 bf16 tile of W^T where W is row-major [N][K] (ld = K).
// lane<16: col n=lane, K = {0..7} and {16..23}; lane>=16: K = {8..15} and {24..31}
__device__ __forceinline__ v16bf load_b_frag(const ushort_t* base, int ld,
                                             int n0, int lane) {
  const int n = lane & 15, kg = lane >> 4;
  const ushort_t* p = base + (n0 + n) * ld + kg * 8;
  FragBits f;
  f.q[0] = *(const u32x4*)(p);
  f.q[1] = *(const u32x4*)(p + 16);
  return f.v;
}

__device__ __forceinline__ v8f wmma_bf16(v16bf a, v16bf b, v8f c) {
  return __builtin_amdgcn_wmma_f32_16x16x32_bf16(false, a, false, b,
                                                 (short)0, c, false, false);
}

// ---- CDNA5 async copy: global -> LDS, 16 B per lane, tracked by ASYNCcnt ----
__device__ __forceinline__ unsigned lds_off(const void* p) {
  // ISA: LDS aperture keeps the LDS byte offset in addr[31:0]
  return (unsigned)(unsigned long long)p;
}

__device__ __forceinline__ void async_ld_b128(unsigned ldsOff, const ushort_t* g) {
  asm volatile("global_load_async_to_lds_b128 %0, %1, off"
               :: "v"(ldsOff), "v"(g)
               : "memory");
}

__device__ __forceinline__ void wait_async0() {
  asm volatile("s_wait_asynccnt 0" ::: "memory");
}

// Stage NR rows x kKC cols of bf16 from global row-major (ld elems) into an LDS
// tile with kRowE element stride. 256 threads: 4 x 16B chunks per row, 64 rows/pass.
template <int NR>
__device__ __forceinline__ void stage_tile(unsigned ldsBase, const ushort_t* g,
                                           int ld, int row0, int k0, int tid) {
  const int r = tid >> 2;       // 0..63
  const int c = tid & 3;        // 16 B chunk within the 64 B row
#pragma unroll
  for (int p = 0; p < NR; p += 64) {
    const int row = p + r;
    async_ld_b128(ldsBase + (unsigned)(row * kRowB + c * 16),
                  g + (long long)(row0 + row) * ld + k0 + c * 8);
  }
}

// ---------------- routing path ----------------

__global__ __launch_bounds__(256) void mean_x_kernel(const float* __restrict__ x,
                                                     float* __restrict__ reduced) {
  const int idx = blockIdx.x * 256 + threadIdx.x;   // 0..kB*kD-1
  const int b = idx >> 10, dd = idx & (kD - 1);
  const float* p = x + (long long)b * kS * kD + dd;
  float s = 0.f;
  for (int i = 0; i < kS; ++i) s += p[(long long)i * kD];
  reduced[idx] = s * (1.0f / kS);
}

__global__ __launch_bounds__(256) void router_kernel(const float* __restrict__ reduced,
                                                     const float* __restrict__ rw,
                                                     const float* __restrict__ rb,
                                                     float* __restrict__ probs) {
  __shared__ float logits[kB * kE];
  const int lane = threadIdx.x & 31;
  const int w = threadIdx.x >> 5;
  for (int p = w; p < kB * kE; p += 8) {
    const int b = p >> 3, e = p & 7;
    float s = 0.f;
    for (int i = lane; i < kD; i += 32)
      s += reduced[b * kD + i] * rw[e * kD + i];
#pragma unroll
    for (int off = 16; off > 0; off >>= 1) s += __shfl_down(s, off, 32);
    if (lane == 0) logits[p] = s + rb[e];
  }
  __syncthreads();
  if (threadIdx.x == 0) {
    float pr[kE];
#pragma unroll
    for (int e = 0; e < kE; ++e) pr[e] = 0.f;
    for (int b = 0; b < kB; ++b) {
      float mx = -3.4e38f;
      for (int e = 0; e < kE; ++e) mx = fmaxf(mx, logits[b * kE + e]);
      float den = 0.f, ex[kE];
      for (int e = 0; e < kE; ++e) { ex[e] = __expf(logits[b * kE + e] - mx); den += ex[e]; }
      const float inv = 1.0f / den;
      for (int e = 0; e < kE; ++e) pr[e] += ex[e] * inv;
    }
    for (int e = 0; e < kE; ++e) probs[e] = pr[e] * (1.0f / kB);
  }
}

// ---------------- casts & weight merging ----------------

__global__ __launch_bounds__(256) void cvt_bf16_kernel(const float* __restrict__ src,
                                                       ushort_t* __restrict__ dst,
                                                       long long n) {
  long long i = (long long)blockIdx.x * 256 + threadIdx.x;
  const long long stride = (long long)gridDim.x * 256;
  for (; i < n; i += stride) dst[i] = f2bf(src[i]);
}

__global__ __launch_bounds__(256) void merge_w_kernel(const float* __restrict__ ew,
                                                      const float* __restrict__ probs,
                                                      ushort_t* __restrict__ out,
                                                      long long n) {
  float pr[kE];
#pragma unroll
  for (int e = 0; e < kE; ++e) pr[e] = probs[e];
  long long i = (long long)blockIdx.x * 256 + threadIdx.x;
  const long long stride = (long long)gridDim.x * 256;
  for (; i < n; i += stride) {
    float s = 0.f;
#pragma unroll
    for (int e = 0; e < kE; ++e) s += pr[e] * ew[(long long)e * n + i];
    out[i] = f2bf(s);
  }
}

__global__ __launch_bounds__(256) void merge_bias_kernel(const float* __restrict__ eb,
                                                         const float* __restrict__ probs,
                                                         float* __restrict__ out, int n) {
  const int i = blockIdx.x * 256 + threadIdx.x;
  if (i >= n) return;
  float s = 0.f;
#pragma unroll
  for (int e = 0; e < kE; ++e) s += probs[e] * eb[e * n + i];
  out[i] = s;
}

// ---------------- GEMM1 (x @ W_up^T) fused bias + GLU, LDS double-buffered ----------------
// Block: 256 threads = 8 waves. Tile: M=128, N_lin=64 (+64 gated cols). K staged 32/chunk.
__global__ __launch_bounds__(256) void gemm1_glu_kernel(const ushort_t* __restrict__ X,   // [kT][kD]
                                                        const ushort_t* __restrict__ Wup, // [kH][kD]
                                                        const float* __restrict__ bup,    // [kH]
                                                        ushort_t* __restrict__ Sout) {    // [kT][kF]
  __shared__ ushort_t smem[4 * 128 * kRowE];   // A dbuf (2x128 rows) + B dbuf (2x128 rows)
  const int tid = threadIdx.x;
  const int lane = tid & 31;
  const int w = tid >> 5;
  const int wm = w & 3, wn = w >> 2;
  const int mBlock = blockIdx.x * 128;
  const int nBlock = blockIdx.y * 64;
  const int mBase = mBlock + wm * 32;
  const int nBase = nBlock + wn * 32;

  constexpr int kTileE = 128 * kRowE;          // elems per 128-row tile
  constexpr unsigned kTileBytes = (unsigned)(kTileE * 2);
  const unsigned sbase = lds_off(smem);

  const v8f vz = {0.f, 0.f, 0.f, 0.f, 0.f, 0.f, 0.f, 0.f};
  v8f accL[2][2], accG[2][2];
#pragma unroll
  for (int i = 0; i < 2; ++i)
#pragma unroll
    for (int j = 0; j < 2; ++j) { accL[i][j] = vz; accG[i][j] = vz; }

  // prologue: stage k=0 into buffer 0 (B tile rows 0..63 = linear, 64..127 = gated)
  {
    const unsigned aOff = sbase;
    const unsigned bOff = sbase + 2u * kTileBytes;
    stage_tile<128>(aOff, X, kD, mBlock, 0, tid);
    stage_tile<64>(bOff, Wup, kD, nBlock, 0, tid);
    stage_tile<64>(bOff + 64u * kRowB, Wup, kD, nBlock + kF, 0, tid);
  }
  wait_async0();
  __syncthreads();

  constexpr int nK = kD / kKC;   // 32
#pragma unroll 1
  for (int kt = 0; kt < nK; ++kt) {
    const int cur = kt & 1;
    const int nxt = cur ^ 1;
    if (kt + 1 < nK) {
      const int k0 = (kt + 1) * kKC;
      const unsigned aOff = sbase + (unsigned)nxt * kTileBytes;
      const unsigned bOff = sbase + 2u * kTileBytes + (unsigned)nxt * kTileBytes;
      stage_tile<128>(aOff, X, kD, mBlock, k0, tid);
      stage_tile<64>(bOff, Wup, kD, nBlock, k0, tid);
      stage_tile<64>(bOff + 64u * kRowB, Wup, kD, nBlock + kF, k0, tid);
    }
    const ushort_t* As = smem + cur * kTileE;
    const ushort_t* Bs = smem + 2 * kTileE + cur * kTileE;
    v16bf a0  = load_a_frag(As, kRowE, wm * 32, lane);
    v16bf a1  = load_a_frag(As, kRowE, wm * 32 + 16, lane);
    v16bf bl0 = load_b_frag(Bs, kRowE, wn * 32, lane);
    v16bf bl1 = load_b_frag(Bs, kRowE, wn * 32 + 16, lane);
    v16bf bg0 = load_b_frag(Bs, kRowE, 64 + wn * 32, lane);
    v16bf bg1 = load_b_frag(Bs, kRowE, 64 + wn * 32 + 16, lane);
    accL[0][0] = wmma_bf16(a0, bl0, accL[0][0]);
    accL[0][1] = wmma_bf16(a0, bl1, accL[0][1]);
    accL[1][0] = wmma_bf16(a1, bl0, accL[1][0]);
    accL[1][1] = wmma_bf16(a1, bl1, accL[1][1]);
    accG[0][0] = wmma_bf16(a0, bg0, accG[0][0]);
    accG[0][1] = wmma_bf16(a0, bg1, accG[0][1]);
    accG[1][0] = wmma_bf16(a1, bg0, accG[1][0]);
    accG[1][1] = wmma_bf16(a1, bg1, accG[1][1]);
    wait_async0();
    __syncthreads();
  }

  const int n = lane & 15, kg = lane >> 4;
#pragma unroll
  for (int i = 0; i < 2; ++i)
#pragma unroll
    for (int j = 0; j < 2; ++j) {
      const int col = nBase + j * 16 + n;
      const float bl = bup[col];
      const float bg = bup[col + kF];
#pragma unroll
      for (int r = 0; r < 8; ++r) {
        const float lin = accL[i][j][r] + bl;
        const float gt = accG[i][j][r] + bg;
        const float sv = lin * (gt / (1.0f + __expf(-gt)));  // lin * silu(gate)
        const int row = mBase + i * 16 + kg * 8 + r;
        Sout[(long long)row * kF + col] = f2bf(sv);
      }
    }
}

// ---------------- GEMM2 (s @ W_dn^T) + bias -> f32, LDS double-buffered ----------------
__global__ __launch_bounds__(256) void gemm2_kernel(const ushort_t* __restrict__ S,   // [kT][kF]
                                                    const ushort_t* __restrict__ Wdn, // [kD][kF]
                                                    const float* __restrict__ bdn,    // [kD]
                                                    float* __restrict__ Out) {        // [kT][kD]
  __shared__ ushort_t smem[2 * 128 * kRowE + 2 * 64 * kRowE];  // A dbuf + B dbuf
  const int tid = threadIdx.x;
  const int lane = tid & 31;
  const int w = tid >> 5;
  const int wm = w & 3, wn = w >> 2;
  const int mBlock = blockIdx.x * 128;
  const int nBlock = blockIdx.y * 64;
  const int mBase = mBlock + wm * 32;
  const int nBase = nBlock + wn * 32;

  constexpr int kATileE = 128 * kRowE;
  constexpr int kBTileE = 64 * kRowE;
  constexpr unsigned kATileBytes = (unsigned)(kATileE * 2);
  constexpr unsigned kBTileBytes = (unsigned)(kBTileE * 2);
  const unsigned sbase = lds_off(smem);

  const v8f vz = {0.f, 0.f, 0.f, 0.f, 0.f, 0.f, 0.f, 0.f};
  v8f acc[2][2];
#pragma unroll
  for (int i = 0; i < 2; ++i)
#pragma unroll
    for (int j = 0; j < 2; ++j) acc[i][j] = vz;

  {
    const unsigned aOff = sbase;
    const unsigned bOff = sbase + 2u * kATileBytes;
    stage_tile<128>(aOff, S, kF, mBlock, 0, tid);
    stage_tile<64>(bOff, Wdn, kF, nBlock, 0, tid);
  }
  wait_async0();
  __syncthreads();

  constexpr int nK = kF / kKC;   // 128
#pragma unroll 1
  for (int kt = 0; kt < nK; ++kt) {
    const int cur = kt & 1;
    const int nxt = cur ^ 1;
    if (kt + 1 < nK) {
      const int k0 = (kt + 1) * kKC;
      const unsigned aOff = sbase + (unsigned)nxt * kATileBytes;
      const unsigned bOff = sbase + 2u * kATileBytes + (unsigned)nxt * kBTileBytes;
      stage_tile<128>(aOff, S, kF, mBlock, k0, tid);
      stage_tile<64>(bOff, Wdn, kF, nBlock, k0, tid);
    }
    const ushort_t* As = smem + cur * kATileE;
    const ushort_t* Bs = smem + 2 * kATileE + cur * kBTileE;
    v16bf a0 = load_a_frag(As, kRowE, wm * 32, lane);
    v16bf a1 = load_a_frag(As, kRowE, wm * 32 + 16, lane);
    v16bf b0 = load_b_frag(Bs, kRowE, wn * 32, lane);
    v16bf b1 = load_b_frag(Bs, kRowE, wn * 32 + 16, lane);
    acc[0][0] = wmma_bf16(a0, b0, acc[0][0]);
    acc[0][1] = wmma_bf16(a0, b1, acc[0][1]);
    acc[1][0] = wmma_bf16(a1, b0, acc[1][0]);
    acc[1][1] = wmma_bf16(a1, b1, acc[1][1]);
    wait_async0();
    __syncthreads();
  }

  const int n = lane & 15, kg = lane >> 4;
#pragma unroll
  for (int i = 0; i < 2; ++i)
#pragma unroll
    for (int j = 0; j < 2; ++j) {
      const int col = nBase + j * 16 + n;
      const float bias = bdn[col];
#pragma unroll
      for (int r = 0; r < 8; ++r) {
        const int row = mBase + i * 16 + kg * 8 + r;
        Out[(long long)row * kD + col] = acc[i][j][r] + bias;
      }
    }
}

extern "C" void kernel_launch(void* const* d_in, const int* in_sizes, int n_in,
                              void* d_out, int out_size, void* d_ws, size_t ws_size,
                              hipStream_t stream) {
  const float* x   = (const float*)d_in[0];
  const float* rw  = (const float*)d_in[1];
  const float* rb  = (const float*)d_in[2];
  const float* upw = (const float*)d_in[3];
  const float* upb = (const float*)d_in[4];
  const float* dnw = (const float*)d_in[5];
  const float* dnb = (const float*)d_in[6];
  float* out = (float*)d_out;

  // workspace carve: f32 scalars/vectors, then 256B-aligned bf16 buffers
  char* ws = (char*)d_ws;
  float* probs   = (float*)ws;            // [8]
  float* reduced = probs + kE;            // [kB*kD]
  float* bup     = reduced + kB * kD;     // [kH]
  float* bdn     = bup + kH;              // [kD]
  size_t off = ((size_t)(kE + kB * kD + kH + kD) * sizeof(float) + 255) & ~(size_t)255;
  ushort_t* xb  = (ushort_t*)(ws + off);  off += (size_t)kT * kD * 2;  // x in bf16
  ushort_t* wup = (ushort_t*)(ws + off);  off += (size_t)kH * kD * 2;  // merged W_up bf16
  ushort_t* wdn = (ushort_t*)(ws + off);  off += (size_t)kD * kF * 2;  // merged W_dn bf16
  ushort_t* sb  = (ushort_t*)(ws + off);                               // s activations bf16
  (void)ws_size; (void)in_sizes; (void)n_in; (void)out_size;

  mean_x_kernel<<<(kB * kD) / 256, 256, 0, stream>>>(x, reduced);
  router_kernel<<<1, 256, 0, stream>>>(reduced, rw, rb, probs);
  cvt_bf16_kernel<<<4096, 256, 0, stream>>>(x, xb, (long long)kT * kD);
  merge_w_kernel<<<4096, 256, 0, stream>>>(upw, probs, wup, (long long)kH * kD);
  merge_w_kernel<<<2048, 256, 0, stream>>>(dnw, probs, wdn, (long long)kD * kF);
  merge_bias_kernel<<<kH / 256, 256, 0, stream>>>(upb, probs, bup, kH);
  merge_bias_kernel<<<kD / 256, 256, 0, stream>>>(dnb, probs, bdn, kD);

  gemm1_glu_kernel<<<dim3(kT / 128, kF / 64), 256, 0, stream>>>(xb, wup, bup, sb);
  gemm2_kernel<<<dim3(kT / 128, kD / 64), 256, 0, stream>>>(sb, wdn, bdn, out);
}
